// DSLOModel_76570676953182
// MI455X (gfx1250) — matrix operand order repaired and checked by reading, not compiled, not run on hardware
//
#include <hip/hip_runtime.h>
#include <hip/hip_bf16.h>

// ---------------------------------------------------------------------------
// Problem constants (from the reference)
// ---------------------------------------------------------------------------
#define BATCH 2048
#define NPTS  360
#define KNN_K 8
#define NDOWN 90      // N // DS
#define HID   192     // lstm hidden
#define G4    768     // 4 * HID

typedef __attribute__((ext_vector_type(16))) __bf16 v16bf;
typedef __attribute__((ext_vector_type(8)))  float  v8f;

__device__ __forceinline__ float sigmoidf_(float x) {
    return 1.0f / (1.0f + __expf(-x));
}

// ---------------------------------------------------------------------------
// KNN aggregate: per batch row, for the 90 down-sampled query points compute
// mean of the 8 nearest 1-D neighbours (self included, stable tie-break).
// Row staged in LDS; top-8 kept in registers (stable insertion sort).
// ---------------------------------------------------------------------------
__global__ void knn_agg_kernel(const float* __restrict__ x,
                               float* __restrict__ agg) {
    __shared__ float xs[NPTS];
    const int b = blockIdx.x;
    const float* xr = x + (size_t)b * NPTS;
    for (int i = threadIdx.x; i < NPTS; i += blockDim.x) xs[i] = xr[i];
    __syncthreads();

    const int t = threadIdx.x;
    if (t < NDOWN) {
        const float xi = xs[4 * t];
        float dist[KNN_K], val[KNN_K];
#pragma unroll
        for (int s = 0; s < KNN_K; ++s) { dist[s] = 3.4e38f; val[s] = 0.0f; }
        for (int j = 0; j < NPTS; ++j) {
            const float xj = xs[j];
            const float d  = fabsf(xi - xj);
            if (d < dist[KNN_K - 1]) {          // strict < keeps lower index on ties
                dist[KNN_K - 1] = d; val[KNN_K - 1] = xj;
#pragma unroll
                for (int s = KNN_K - 1; s > 0; --s) {
                    if (dist[s] < dist[s - 1]) {
                        float td = dist[s]; dist[s] = dist[s - 1]; dist[s - 1] = td;
                        float tv = val[s];  val[s]  = val[s - 1];  val[s - 1]  = tv;
                    }
                }
            }
        }
        float m = 0.0f;
#pragma unroll
        for (int s = 0; s < KNN_K; ++s) m += val[s];
        agg[(size_t)b * NDOWN + t] = m * (1.0f / KNN_K);
    }
}

// ---------------------------------------------------------------------------
// WMMA GEMM:  out(M,N) = act( A(M,K) * W(N,K)^T + bias1 + bias2 )
// One wave computes a 16x64 strip of C: one shared A fragment is reused
// across 4 N-tiles (4 back-to-back v_wmma_f32_16x16x32_bf16 per 32-K step,
// amortizing A loads + bf16 conversion 4x and pipelining the matrix unit).
// fp32 inputs converted to bf16 in-flight; fp32 accumulation. Tail-K handled
// with zero-padded loads so EXEC stays all-1s at the WMMA.
// MODE: 0=linear 1=relu 2=tanh(relu) 3=sigmoid
// Requires N % 64 == 0, M % 16 == 0 (true for all call sites).
// ---------------------------------------------------------------------------
template <int MODE>
__global__ void gemm_wmma_bf16(const float* __restrict__ A,
                               const float* __restrict__ W,
                               const float* __restrict__ bias1,
                               const float* __restrict__ bias2,
                               float* __restrict__ out,
                               int M, int N, int K) {
    const int lane = threadIdx.x;       // 0..31, one wave per block
    const int hf   = lane >> 4;         // half-wave select
    const int l16  = lane & 15;

    const float* pA = A + (size_t)(blockIdx.y * 16 + l16) * K;          // A row
    const float* pW[4];
#pragma unroll
    for (int t = 0; t < 4; ++t) {
        pW[t] = W + (size_t)(blockIdx.x * 64 + t * 16 + l16) * K;       // W rows
        __builtin_prefetch(pW[t], 0, 3);                                // global_prefetch_b8
    }

    v8f acc[4] = {};
    for (int k0 = 0; k0 < K; k0 += 32) {
        v16bf afrag;
        v16bf bfrag[4];
        if (k0 + 32 <= K) {
            // A 16x32 bf16 layout: lanes 0-15 K=0..15, lanes 16-31 K=16..31
#pragma unroll
            for (int j = 0; j < 8; ++j)
                afrag[j] = (__bf16)pA[k0 + hf * 16 + j];
#pragma unroll
            for (int j = 0; j < 8; ++j)
                afrag[8 + j] = (__bf16)pA[k0 + hf * 16 + 8 + j];
            // B 32x16 bf16 layout: lanes 0-15 K=0..15, lanes 16-31 K=16..31
#pragma unroll
            for (int t = 0; t < 4; ++t)
#pragma unroll
                for (int j = 0; j < 16; ++j)
                    bfrag[t][j] = (__bf16)pW[t][k0 + hf * 16 + j];
        } else {
            // zero-padded tail (exact: zeros contribute nothing)
#pragma unroll
            for (int j = 0; j < 8; ++j) {
                int ka = k0 + hf * 16 + j;
                afrag[j] = (__bf16)((ka < K) ? pA[ka] : 0.0f);
            }
#pragma unroll
            for (int j = 0; j < 8; ++j) {
                int ka = k0 + hf * 16 + 8 + j;
                afrag[8 + j] = (__bf16)((ka < K) ? pA[ka] : 0.0f);
            }
#pragma unroll
            for (int t = 0; t < 4; ++t)
#pragma unroll
                for (int j = 0; j < 16; ++j) {
                    int kb = k0 + hf * 16 + j;
                    bfrag[t][j] = (__bf16)((kb < K) ? pW[t][kb] : 0.0f);
                }
        }
#pragma unroll
        for (int t = 0; t < 4; ++t)
            acc[t] = __builtin_amdgcn_wmma_f32_16x16x32_bf16(
                /*neg_a=*/false, afrag, /*neg_b=*/false, bfrag[t],
                /*c_mod=*/(short)0, acc[t], /*reuse_a=*/false, /*reuse_b=*/false);
    }

    // Epilogue: C/D layout: VGPR v -> row v (lanes 0-15) / row v+8 (lanes 16-31)
#pragma unroll
    for (int t = 0; t < 4; ++t) {
        const int col = blockIdx.x * 64 + t * 16 + l16;
        float bsum = 0.0f;
        if (bias1) bsum += bias1[col];
        if (bias2) bsum += bias2[col];
#pragma unroll
        for (int v = 0; v < 8; ++v) {
            const int row = blockIdx.y * 16 + hf * 8 + v;
            float xv = acc[t][v] + bsum;
            if (MODE == 1) xv = fmaxf(xv, 0.0f);
            if (MODE == 2) xv = tanhf(fmaxf(xv, 0.0f));
            if (MODE == 3) xv = sigmoidf_(xv);
            out[(size_t)row * N + col] = xv;
        }
    }
}

// ---------------------------------------------------------------------------
// LSTM gates (h0=c0=0): c = sig(i)*tanh(g); h = sig(o)*tanh(c)
// Gate order i,f,g,o at columns j, j+H, j+2H, j+3H of the 4H-wide pre-act.
// Optionally writes c of the LAST batch row (needed for the peephole quirk).
// ---------------------------------------------------------------------------
__global__ void lstm_gates_kernel(const float* __restrict__ g,
                                  float* __restrict__ h,
                                  float* __restrict__ cLastOut) {
    const int tid = blockIdx.x * blockDim.x + threadIdx.x;
    if (tid >= BATCH * HID) return;
    const int b = tid / HID, j = tid % HID;
    const float* gr = g + (size_t)b * G4;
    const float ig = gr[j];
    const float gg = gr[2 * HID + j];
    const float og = gr[3 * HID + j];
    const float c  = sigmoidf_(ig) * tanhf(gg);
    h[tid] = sigmoidf_(og) * tanhf(c);
    if (cLastOut != nullptr && b == BATCH - 1) cLastOut[j] = c;
}

// ---------------------------------------------------------------------------
// Peephole temporal: uses h1 (B,H) and c_last = c1[last row] (H,)
// ---------------------------------------------------------------------------
__global__ void temporal_kernel(const float* __restrict__ h1,
                                const float* __restrict__ cLast,
                                const float* __restrict__ pwf,
                                const float* __restrict__ pwi,
                                const float* __restrict__ pwo,
                                float* __restrict__ temporal) {
    const int tid = blockIdx.x * blockDim.x + threadIdx.x;
    if (tid >= BATCH * HID) return;
    const int j = tid % HID;
    const float hl = h1[tid], cl = cLast[j];
    const float f_g  = sigmoidf_(hl + pwf[j] * cl);
    const float i_g  = sigmoidf_(hl + pwi[j] * cl);
    const float o_g  = sigmoidf_(hl + pwo[j] * cl);
    const float cell = i_g * tanhf(cl) + f_g * cl;
    temporal[tid] = o_g * tanhf(cell);
}

__global__ void gated_kernel(const float* __restrict__ temporal,
                             const float* __restrict__ z,
                             const float* __restrict__ r,
                             float* __restrict__ gated) {
    const int tid = blockIdx.x * blockDim.x + threadIdx.x;
    if (tid >= BATCH * HID) return;
    gated[tid] = r[tid] * temporal[tid] * z[tid];
}

// Final projection (B,H) x (3,H)^T + bp -> (B,3). Tiny; one thread per row.
__global__ void proj_kernel(const float* __restrict__ gated,
                            const float* __restrict__ wp,
                            const float* __restrict__ bp,
                            float* __restrict__ out) {
    const int row = blockIdx.x * blockDim.x + threadIdx.x;
    if (row >= BATCH) return;
    const float* gr = gated + (size_t)row * HID;
    float a0 = bp[0], a1 = bp[1], a2 = bp[2];
    for (int k = 0; k < HID; ++k) {
        const float t = gr[k];
        a0 += t * wp[k];
        a1 += t * wp[HID + k];
        a2 += t * wp[2 * HID + k];
    }
    out[row * 3 + 0] = a0;
    out[row * 3 + 1] = a1;
    out[row * 3 + 2] = a2;
}

// ---------------------------------------------------------------------------
// Launcher
// ---------------------------------------------------------------------------
extern "C" void kernel_launch(void* const* d_in, const int* in_sizes, int n_in,
                              void* d_out, int out_size, void* d_ws, size_t ws_size,
                              hipStream_t stream) {
    (void)in_sizes; (void)n_in; (void)out_size; (void)ws_size;

    const float* lidar = (const float*)d_in[0];
    const float* w1a   = (const float*)d_in[1];
    const float* b1a   = (const float*)d_in[2];
    const float* w1b   = (const float*)d_in[3];
    const float* b1b   = (const float*)d_in[4];
    const float* w2    = (const float*)d_in[5];
    const float* b2    = (const float*)d_in[6];
    const float* wih0  = (const float*)d_in[7];
    // d_in[8] = whh0 (unused: h0=0)
    const float* bih0  = (const float*)d_in[9];
    const float* bhh0  = (const float*)d_in[10];
    const float* wih1  = (const float*)d_in[11];
    // d_in[12] = whh1 (unused)
    const float* bih1  = (const float*)d_in[13];
    const float* bhh1  = (const float*)d_in[14];
    const float* pwf   = (const float*)d_in[15];
    const float* pwi   = (const float*)d_in[16];
    const float* pwo   = (const float*)d_in[17];
    const float* wz    = (const float*)d_in[18];
    const float* bz    = (const float*)d_in[19];
    const float* wr    = (const float*)d_in[20];
    const float* br    = (const float*)d_in[21];
    const float* wp    = (const float*)d_in[22];
    const float* bp    = (const float*)d_in[23];
    float* out = (float*)d_out;

    // workspace layout (floats)
    float* ws = (float*)d_ws;
    size_t off = 0;
    float* agg      = ws + off; off += (size_t)BATCH * NDOWN;  // (B,90)
    float* h1a      = ws + off; off += (size_t)BATCH * 256;    // (B,256)
    float* x2       = ws + off; off += (size_t)BATCH * 128;    // (B,128)
    float* feat     = ws + off; off += (size_t)BATCH * 128;    // (B,128)
    float* g0       = ws + off; off += (size_t)BATCH * G4;     // (B,768)
    float* h0       = ws + off; off += (size_t)BATCH * HID;    // (B,192)
    float* g1       = ws + off; off += (size_t)BATCH * G4;     // (B,768)
    float* h1       = ws + off; off += (size_t)BATCH * HID;    // (B,192)
    float* cLast    = ws + off; off += HID;                    // (192,)
    float* temporal = ws + off; off += (size_t)BATCH * HID;    // (B,192)
    float* zbuf     = ws + off; off += (size_t)BATCH * HID;    // (B,192)
    float* rbuf     = ws + off; off += (size_t)BATCH * HID;    // (B,192)
    float* gated    = ws + off; off += (size_t)BATCH * HID;    // (B,192)

    // 1) KNN aggregate -> (B,90)
    knn_agg_kernel<<<BATCH, 128, 0, stream>>>(lidar, agg);

    const dim3 wave(32, 1, 1);
    // 2) (B,90)x(256,90)^T  relu
    gemm_wmma_bf16<1><<<dim3(256 / 64, BATCH / 16), wave, 0, stream>>>(
        agg, w1a, b1a, nullptr, h1a, BATCH, 256, NDOWN);
    // 3) (B,256)x(128,256)^T relu
    gemm_wmma_bf16<1><<<dim3(128 / 64, BATCH / 16), wave, 0, stream>>>(
        h1a, w1b, b1b, nullptr, x2, BATCH, 128, 256);
    // 4) (B,128)x(128,128)^T tanh(relu)
    gemm_wmma_bf16<2><<<dim3(128 / 64, BATCH / 16), wave, 0, stream>>>(
        x2, w2, b2, nullptr, feat, BATCH, 128, 128);
    // 5) LSTM layer 0 pre-activations (B,768) = feat x wih0^T + bih0 + bhh0
    gemm_wmma_bf16<0><<<dim3(G4 / 64, BATCH / 16), wave, 0, stream>>>(
        feat, wih0, bih0, bhh0, g0, BATCH, G4, 128);
    // 6) gates -> h0
    lstm_gates_kernel<<<(BATCH * HID + 255) / 256, 256, 0, stream>>>(g0, h0, nullptr);
    // 7) LSTM layer 1 pre-activations
    gemm_wmma_bf16<0><<<dim3(G4 / 64, BATCH / 16), wave, 0, stream>>>(
        h0, wih1, bih1, bhh1, g1, BATCH, G4, HID);
    // 8) gates -> h1, and c1 of last batch row -> cLast
    lstm_gates_kernel<<<(BATCH * HID + 255) / 256, 256, 0, stream>>>(g1, h1, cLast);
    // 9) peephole temporal
    temporal_kernel<<<(BATCH * HID + 255) / 256, 256, 0, stream>>>(
        h1, cLast, pwf, pwi, pwo, temporal);
    // 10) z / r gates, sigmoid epilogue (N=192 = 3 strips of 64)
    gemm_wmma_bf16<3><<<dim3(HID / 64, BATCH / 16), wave, 0, stream>>>(
        temporal, wz, bz, nullptr, zbuf, BATCH, HID, HID);
    gemm_wmma_bf16<3><<<dim3(HID / 64, BATCH / 16), wave, 0, stream>>>(
        temporal, wr, br, nullptr, rbuf, BATCH, HID, HID);
    // 11) gated combine
    gated_kernel<<<(BATCH * HID + 255) / 256, 256, 0, stream>>>(
        temporal, zbuf, rbuf, gated);
    // 12) final projection -> (B,3)
    proj_kernel<<<(BATCH + 255) / 256, 256, 0, stream>>>(gated, wp, bp, out);
}
